// GENI_27960237097211
// MI455X (gfx1250) — matrix operand order repaired and checked by reading, not compiled, not run on hardware
//
#include <hip/hip_runtime.h>
#include <stdint.h>

#define NN 4096
#define TT 3
#define DD 20
#define EE 131072
#define ATT_DIMC 128
#define HEADSC 2
#define NEG_SLOPE_C 0.2f

typedef __attribute__((ext_vector_type(16))) _Float16 v16h;
typedef __attribute__((ext_vector_type(8)))  float    v8f;
typedef __attribute__((ext_vector_type(4)))  unsigned int u32x4;
typedef __attribute__((ext_vector_type(8)))  int          i32x8;
typedef __attribute__((ext_vector_type(4)))  int          i32x4;

// ---------------------------------------------------------------------------
// TDM: issue an async DMA of a 16-row x 256-byte tile (row stride 4096 B)
// from global `gaddr` into LDS byte-address `laddr`.  D# packed per CDNA5 ISA
// sec 8.3/8.4.  Completion tracked with TENSORcnt.
// ---------------------------------------------------------------------------
__device__ __forceinline__ unsigned int lds_addr_of(const void* p) {
  // low 32 bits of the flat shared-aperture address == LDS byte address
  return (unsigned int)(uintptr_t)p;
}

__device__ __forceinline__ void tdm_load_tile(const void* gaddr, unsigned int laddr) {
  const uint64_t ga = (uint64_t)(uintptr_t)gaddr;
  u32x4 g0;
  g0[0] = 1u;                                   // count=1 valid user descriptor
  g0[1] = laddr;                                // lds_addr
  g0[2] = (unsigned int)ga;                     // global_addr[31:0]
  g0[3] = (unsigned int)((ga >> 32) & 0x01FFFFFFu) | (2u << 30); // ga[56:32] | type=2
  i32x8 g1;
  g1[0] = 0;                                    // wg_mask=0, data_size=1B, no flags
  g1[1] = (int)(4096u << 16);                   // tensor_dim0[15:0]=4096
  g1[2] = (int)(4096u << 16);                   // tensor_dim0 hi=0 | tensor_dim1 lo=4096
  g1[3] = (int)(256u << 16);                    // tensor_dim1 hi=0 | tile_dim0=256
  g1[4] = 16;                                   // tile_dim1=16, tile_dim2=0
  g1[5] = 4096;                                 // tensor_dim0_stride = 4096
  g1[6] = 0;                                    // stride hi | tensor_dim1_stride lo
  g1[7] = 0;                                    // tensor_dim1_stride hi
  i32x4 g2;                                     // benign 3rd/4th dims
  g2[0] = 1; g2[1] = 1; g2[2] = 0; g2[3] = 0;   // tensor_dim2=1, tensor_dim3=1
  i32x4 g3;
  g3[0] = 0; g3[1] = (int)(1u << 16);           // tensor_dim4=1
  g3[2] = 0; g3[3] = 0;
#if __clang_major__ >= 23
  i32x8 z8;
  z8[0]=0; z8[1]=0; z8[2]=0; z8[3]=0; z8[4]=0; z8[5]=0; z8[6]=0; z8[7]=0;
  __builtin_amdgcn_tensor_load_to_lds(g0, g1, g2, g3, z8, 0);
#else
  __builtin_amdgcn_tensor_load_to_lds(g0, g1, g2, g3, 0);
#endif
}

// ---------------------------------------------------------------------------
// Kernel 1: scatter edge lists into a 3-bit-per-pair adjacency mask.
// atomicOr is idempotent -> duplicates/order don't matter (deterministic,
// matches jnp .set(1.0) semantics).
// ---------------------------------------------------------------------------
__global__ void build_mask_kernel(const int* __restrict__ el,
                                  unsigned int* __restrict__ mask32) {
  unsigned idx = blockIdx.x * blockDim.x + threadIdx.x;
  if (idx >= TT * EE) return;
  unsigned t = idx / EE;
  unsigned e = idx - t * EE;
  unsigned src = (unsigned)el[(t * 2 + 0) * EE + e] & (NN - 1);
  unsigned tgt = (unsigned)el[(t * 2 + 1) * EE + e] & (NN - 1);
  unsigned p0 = src * NN + tgt;
  unsigned p1 = tgt * NN + src;
  atomicOr(&mask32[p0 >> 2], 1u << (((p0 & 3u) << 3) + t));
  atomicOr(&mask32[p1 >> 2], 1u << (((p1 & 3u) << 3) + t));
}

// ---------------------------------------------------------------------------
// Kernel 2: scores = inputs @ lin_w.T + lin_b  (one wave per row, exact f32)
// ---------------------------------------------------------------------------
__global__ void gemv_scores_kernel(const float* __restrict__ x,
                                   const float* __restrict__ w,
                                   const float* __restrict__ b,
                                   float* __restrict__ s) {
  const int gid  = blockIdx.x * blockDim.x + threadIdx.x;
  const int lane = gid & 31;
  const int row  = gid >> 5;
  const float* xr = x + (size_t)row * ATT_DIMC;
  float acc = 0.f;
#pragma unroll
  for (int k = 0; k < ATT_DIMC / 32; ++k)
    acc += xr[lane + 32 * k] * w[lane + 32 * k];
#pragma unroll
  for (int off = 16; off > 0; off >>= 1)
    acc += __shfl_xor(acc, off, 32);
  if (lane == 0) s[row] = acc + b[0];
}

// ---------------------------------------------------------------------------
// Kernel 3: one attention layer (both heads fused; one mask pass).
// Block = 4 waves; wave owns a 16-row tile x 1024-column segment.
// Mask tiles (16x256B, row stride 4096) are DMA'd into LDS by the TDM,
// double-buffered, one tile in flight (s_wait_tensorcnt 1).  Per 16x32 chunk:
// unpack mask bits -> exp(leaky_relu(logit)) in f32, pack f16 A-fragment,
// reduce {sum e, sum e*s_j} with V_WMMA_F32_16X16X32_F16, B = [1 | s_j | 0..],
// f32 accumulators.  Deterministic fixed-order cross-wave combine via LDS.
// ---------------------------------------------------------------------------
__global__ __launch_bounds__(128)
void attn_layer_kernel(const uint8_t* __restrict__ mask,
                       const float* __restrict__ s_in,
                       const float* __restrict__ edge_emb,
                       const float* __restrict__ att_w,
                       int layer,
                       float* __restrict__ s_out) {
  __shared__ __align__(16) uint8_t mbuf[4][2][16 * 256];  // 32 KiB TDM staging
  __shared__ float s_stage[4][32];
  __shared__ float part[4][HEADSC][2][16];

  const int tid  = threadIdx.x;
  const int lane = tid & 31;
  const int wv   = tid >> 5;
  const int tile = blockIdx.x;
  const int m    = lane & 15;
  const int row  = tile * 16 + m;
  const int kb   = (lane >> 4) * 8;   // K-base for A-fragment lane halves

  // Per-head constants (tiny; scalarized by compiler, fully cached)
  float w0[HEADSC], wl[HEADSC], et[HEADSC][TT];
#pragma unroll
  for (int h = 0; h < HEADSC; ++h) {
    const float* w = att_w + (size_t)(layer * HEADSC + h) * (DD + 2);
    w0[h] = w[0];
    wl[h] = w[DD + 1];
#pragma unroll
    for (int t = 0; t < TT; ++t) {
      float a = 0.f;
      for (int d = 0; d < DD; ++d) a += edge_emb[t * DD + d] * w[1 + d];
      et[h][t] = a;
    }
  }

  const float si  = s_in[row];
  const float ci0 = w0[0] * si;
  const float ci1 = w0[1] * si;

  v8f c0, c1;
#pragma unroll
  for (int r = 0; r < 8; ++r) { c0[r] = 0.f; c1[r] = 0.f; }

  const int seg = wv * (NN / 4);
  const uint8_t* gtile = mask + (size_t)(tile * 16) * NN + seg;
  uint8_t* mb0 = &mbuf[wv][0][0];
  uint8_t* mb1 = &mbuf[wv][1][0];

  // prologue: DMA first 16x256 tile
  tdm_load_tile(gtile, lds_addr_of(mb0));

#pragma unroll
  for (int grp = 0; grp < 4; ++grp) {
    if (grp + 1 < 4) {
      tdm_load_tile(gtile + (grp + 1) * 256,
                    lds_addr_of(((grp + 1) & 1) ? mb1 : mb0));
      __builtin_amdgcn_s_wait_tensorcnt(1);   // tile `grp` has landed
    } else {
      __builtin_amdgcn_s_wait_tensorcnt(0);   // last tile has landed
    }
    const uint8_t* mb = (grp & 1) ? mb1 : mb0;

    for (int ch = 0; ch < 8; ++ch) {
      const int j0 = seg + grp * 256 + ch * 32;

      // stage this chunk's s_j (same-wave LDS RAW is in-order on CDNA5)
      s_stage[wv][lane] = s_in[j0 + lane];
      __builtin_amdgcn_wave_barrier();
      if (j0 + 64 <= NN)
        __builtin_prefetch(s_in + j0 + 32, 0, 1);   // global_prefetch_b8

      // mask bytes for this lane's row from the LDS-staged tile
      const uint64_t blo = *(const uint64_t*)(mb + m * 256 + ch * 32 + kb);
      const uint64_t bhi = *(const uint64_t*)(mb + m * 256 + ch * 32 + kb + 16);

      // B fragment (32x16 f16): col0 = ones -> denom; col1 = s_j -> numer
      v16h bf;
#pragma unroll
      for (int i = 0; i < 16; ++i) bf[i] = (_Float16)0.f;
      {
        const int col = lane & 15;
        const int kbB = (lane >> 4) * 16;
        if (col == 0) {
#pragma unroll
          for (int i = 0; i < 16; ++i) bf[i] = (_Float16)1.0f;
        } else if (col == 1) {
#pragma unroll
          for (int i = 0; i < 16; ++i) bf[i] = (_Float16)s_stage[wv][(i < 8) ? (kb + i) : (16 + kb + (i - 8))];
        }
      }

      // A fragments: exp(leaky_relu(logits)) for both heads
      v16h a0, a1;
#pragma unroll
      for (int i = 0; i < 16; ++i) {
        const unsigned bits = (i < 8) ? (unsigned)((blo >> (8 * i)) & 0xFFu)
                                      : (unsigned)((bhi >> (8 * (i - 8))) & 0xFFu);
        const int jj = (i < 8) ? (kb + i) : (16 + kb + (i - 8));
        const float sj  = s_stage[wv][jj];
        const float cnt = (float)__popc(bits);

        float aw0 = ((bits & 1u) ? et[0][0] : 0.f)
                  + ((bits & 2u) ? et[0][1] : 0.f)
                  + ((bits & 4u) ? et[0][2] : 0.f);
        float lg0 = cnt * (ci0 + wl[0] * sj) + aw0;
        lg0 = (lg0 < 0.f) ? (NEG_SLOPE_C * lg0) : lg0;
        const float e0 = bits ? __expf(lg0) : 1.0f;   // exactly 1 for non-edges
        a0[i] = (_Float16)e0;

        float aw1 = ((bits & 1u) ? et[1][0] : 0.f)
                  + ((bits & 2u) ? et[1][1] : 0.f)
                  + ((bits & 4u) ? et[1][2] : 0.f);
        float lg1 = cnt * (ci1 + wl[1] * sj) + aw1;
        lg1 = (lg1 < 0.f) ? (NEG_SLOPE_C * lg1) : lg1;
        const float e1 = bits ? __expf(lg1) : 1.0f;
        a1[i] = (_Float16)e1;
      }

      // EXEC is all-ones here: WMMA legal.
      c0 = __builtin_amdgcn_wmma_f32_16x16x32_f16(false, a0, false, bf,
                                                  (short)0, c0, false, false);
      c1 = __builtin_amdgcn_wmma_f32_16x16x32_f16(false, a1, false, bf,
                                                  (short)0, c1, false, false);
    }
  }

  // D layout: VGPR r, lanes 0-15 = (M=r, N=lane); lanes 16-31 = (M=r+8, N=lane-16).
  // Only columns 0 (denom) and 1 (numer) matter -> lanes {0,1,16,17}.
  if ((lane & 14) == 0) {
    const int col = lane & 15;        // 0 or 1
    const int mb_ = (lane >> 4) * 8;  // rows 0-7 or 8-15
#pragma unroll
    for (int r = 0; r < 8; ++r) {
      part[wv][0][col][mb_ + r] = c0[r];
      part[wv][1][col][mb_ + r] = c1[r];
    }
  }
  __syncthreads();

  if (tid < 16) {
    float d0 = 0.f, n0 = 0.f, d1 = 0.f, n1 = 0.f;
#pragma unroll
    for (int w = 0; w < 4; ++w) {   // fixed order -> deterministic
      d0 += part[w][0][0][tid]; n0 += part[w][0][1][tid];
      d1 += part[w][1][0][tid]; n1 += part[w][1][1][tid];
    }
    s_out[tile * 16 + tid] = 0.5f * (n0 / d0 + n1 / d1);
  }
}

// ---------------------------------------------------------------------------
extern "C" void kernel_launch(void* const* d_in, const int* in_sizes, int n_in,
                              void* d_out, int out_size, void* d_ws, size_t ws_size,
                              hipStream_t stream) {
  (void)in_sizes; (void)n_in; (void)out_size; (void)ws_size;

  const float* inputs   = (const float*)d_in[0];
  const float* lin_w    = (const float*)d_in[1];
  const float* lin_b    = (const float*)d_in[2];
  const float* edge_emb = (const float*)d_in[3];
  const float* att_w    = (const float*)d_in[4];
  const int*   el       = (const int*)d_in[5];

  uint8_t* mask = (uint8_t*)d_ws;                       // N*N bytes = 16 MiB
  float*   s0   = (float*)((char*)d_ws + (size_t)NN * NN);
  float*   s1   = s0 + NN;

  hipMemsetAsync(d_ws, 0, (size_t)NN * NN, stream);     // graph-capture safe

  build_mask_kernel<<<(TT * EE + 255) / 256, 256, 0, stream>>>(
      el, (unsigned int*)d_ws);

  gemv_scores_kernel<<<(NN * 32) / 256, 256, 0, stream>>>(
      inputs, lin_w, lin_b, s0);

  attn_layer_kernel<<<NN / 16, 128, 0, stream>>>(
      mask, s0, edge_emb, att_w, 0, s1);
  attn_layer_kernel<<<NN / 16, 128, 0, stream>>>(
      mask, s1, edge_emb, att_w, 1, (float*)d_out);
}
